// FullAttention_ablation_10144712753260
// MI455X (gfx1250) — matrix operand order repaired and checked by reading, not compiled, not run on hardware
//
#include <hip/hip_runtime.h>
#include <hip/hip_bf16.h>

// ---------------------------------------------------------------------------
// Attention with learned inside-softmax modulation, CDNA5 (gfx1250), wave32.
// B=16, L=S=1024, H=8, E=D=64.
//   scores = (Q K^T) * (1/8) * w ; A = softmax(scores) ; O = A V
// ws layout: wT f16 [S,L] (transposed modulation) | Qh f16 [B,H,L,E]
//            (pre-scaled by 1/8) | Kh f16 [B,H,S,E] | Vt f16 [B,H,D,S]
// K/V WMMA B-fragments are read straight from global (two 16-byte runs, 32 bytes apart, per
// lane -> global_load_b128 pairs; tiles are L2-resident). LDS holds only the
// score matrix + reduction scratch. Cross-lane reductions use ds_swizzle.
// Modulation weights are read as one b128 per C-fragment column (wT layout).
// ---------------------------------------------------------------------------

typedef __attribute__((ext_vector_type(16))) _Float16 v16h;
typedef __attribute__((ext_vector_type(8)))  _Float16 v8h;
typedef __attribute__((ext_vector_type(4)))  _Float16 v4h;
typedef __attribute__((ext_vector_type(8)))  float    v8f;
typedef __attribute__((ext_vector_type(4)))  float    v4f;

namespace cfg {
constexpr int B = 16, H = 8, L = 1024, S = 1024, E = 64, D = 64;
constexpr int M_BLK = 16;      // query rows per block
constexpr int S_BLK = 128;     // keys per tile (8 waves -> 8 col tiles of 16)
constexpr int SS_STR = 1040;   // 1024 + pad halves
constexpr size_t PLANE = (size_t)B * H * L * E;   // 8,388,608 halves per tensor
}

union FragU { v16h v; v8h h[2]; };

// ---- cross-lane reduce via ds_swizzle (xor mode, and_mask=0x1f) -----------
template <int MASK>
__device__ __forceinline__ float swz_max(float x) {
  const int pat = (MASK << 10) | 0x1f;
  return fmaxf(x, __int_as_float(
      __builtin_amdgcn_ds_swizzle(__float_as_int(x), pat)));
}
template <int MASK>
__device__ __forceinline__ float swz_add(float x) {
  const int pat = (MASK << 10) | 0x1f;
  return x + __int_as_float(
      __builtin_amdgcn_ds_swizzle(__float_as_int(x), pat));
}

// A-matrix frag (16-bit 16x32 MxK): lane m=lane&15; lanes<16 hold K
// {kb..kb+7, kb+16..kb+23}; lanes>=16 hold {kb+8..kb+15, kb+24..kb+31}.
__device__ __forceinline__ v16h load_a_frag(const _Float16* __restrict__ base,
                                            int row_base, int stride_h, int k_base, int lane) {
  const int m  = lane & 15;
  const int hi = (lane >> 4) & 1;
  const _Float16* p = base + (size_t)(row_base + m) * stride_h + k_base + hi * 8;
  FragU u;
  u.h[0] = *(const v8h*)(p);
  u.h[1] = *(const v8h*)(p + 16);
  return u.v;
}

// B-matrix frag (16-bit 32x16 KxN): lane n=lane&15; lanes<16 hold K
// {kb..kb+7, kb+16..kb+23}; lanes>=16 hold {kb+8..kb+15, kb+24..kb+31}.
__device__ __forceinline__ v16h load_b_frag(const _Float16* __restrict__ base,
                                            int n_base, int stride_h, int k_base, int lane) {
  const int n  = lane & 15;
  const int hi = (lane >> 4) & 1;
  const _Float16* p = base + (size_t)(n_base + n) * stride_h + k_base + hi * 8;
  FragU u;
  u.h[0] = *(const v8h*)(p);
  u.h[1] = *(const v8h*)(p + 16);
  return u.v;
}

// ---------------------------------------------------------------------------
// Kernel 1: wT[s][row] = softmax_s(weight_mat[row][s] / softplus(tau[row]))
// (stored transposed, f16, so the attention kernel reads contiguous rows)
// ---------------------------------------------------------------------------
__global__ __launch_bounds__(256)
void wprep_kernel(const float* __restrict__ Wm, const float* __restrict__ tau,
                  _Float16* __restrict__ wt) {
  const int row = blockIdx.x;
  const int tid = threadIdx.x;
  __shared__ float red[16];

  const float t   = tau[row];
  const float sp  = (t > 20.f) ? t : log1pf(__expf(t));
  const float inv = 1.0f / sp;
  const float* wr = Wm + (size_t)row * cfg::S;

  float m = -3.0e38f;
  for (int i = tid; i < cfg::S; i += 256) m = fmaxf(m, wr[i] * inv);
  m = swz_max<1>(m); m = swz_max<2>(m); m = swz_max<4>(m);
  m = swz_max<8>(m); m = swz_max<16>(m);
  if ((tid & 31) == 0) red[tid >> 5] = m;
  __syncthreads();
  float bm = red[0];
#pragma unroll
  for (int j = 1; j < 8; ++j) bm = fmaxf(bm, red[j]);

  float s = 0.f;
  for (int i = tid; i < cfg::S; i += 256) s += __expf(wr[i] * inv - bm);
  s = swz_add<1>(s); s = swz_add<2>(s); s = swz_add<4>(s);
  s = swz_add<8>(s); s = swz_add<16>(s);
  if ((tid & 31) == 0) red[8 + (tid >> 5)] = s;
  __syncthreads();
  float st = 0.f;
#pragma unroll
  for (int j = 0; j < 8; ++j) st += red[8 + j];
  const float invs = 1.0f / st;

  for (int i = tid; i < cfg::S; i += 256)
    wt[(size_t)i * cfg::L + row] = (_Float16)(__expf(wr[i] * inv - bm) * invs);
}

// ---------------------------------------------------------------------------
// Kernel 2: layout/precision conversion.
//   plane 0: Qh[b,h,l,e] = q[b,l,h,e] * 0.125 (f16)
//   plane 1: Kh[b,h,s,e] = k[b,s,h,e]         (f16)
//   plane 2: Vt[b,h,d,s] = v[b,s,h,d]         (f16, transposed)
// ---------------------------------------------------------------------------
__global__ __launch_bounds__(256)
void cvt_kernel(const float* __restrict__ q, const float* __restrict__ k,
                const float* __restrict__ v, _Float16* __restrict__ qh,
                _Float16* __restrict__ kh, _Float16* __restrict__ vt) {
  using namespace cfg;
  const int t = blockIdx.x * 256 + threadIdx.x;   // 2,097,152 per plane
  const int plane = blockIdx.y;
  if (plane < 2) {                                 // Q or K reorder
    const int e4 = (t & 15) * 4;
    const int h  = (t >> 4) & 7;
    const int l  = (t >> 7) & 1023;
    const int b  = t >> 17;
    const float sc = (plane == 0) ? 0.125f : 1.0f;
    const float* src = (plane == 0) ? q : k;
    _Float16*  dst = (plane == 0) ? qh : kh;
    v4f x = *(const v4f*)(src + (((size_t)b * L + l) * H + h) * E + e4);
    v4h hv;
#pragma unroll
    for (int u = 0; u < 4; ++u) hv[u] = (_Float16)(x[u] * sc);
    *(v4h*)(dst + (((size_t)b * H + h) * L + l) * E + e4) = hv;
  } else {                                         // V transpose
    const int s4 = (t & 255) * 4;
    const int d  = (t >> 8) & 63;
    const int h  = (t >> 14) & 7;
    const int b  = t >> 17;
    v4h hv;
#pragma unroll
    for (int u = 0; u < 4; ++u)
      hv[u] = (_Float16)v[(((size_t)b * S + s4 + u) * H + h) * D + d];
    *(v4h*)(vt + (((size_t)b * H + h) * D + d) * S + s4) = hv;
  }
}

// ---------------------------------------------------------------------------
// Kernel 3: attention. grid = (L/16, H, B), 256 threads (8 waves).
// ---------------------------------------------------------------------------
__global__ __launch_bounds__(256)
void attn_kernel(const _Float16* __restrict__ qh, const _Float16* __restrict__ kh,
                 const _Float16* __restrict__ vt, const _Float16* __restrict__ wt,
                 float* __restrict__ out) {
  using namespace cfg;
  __shared__ _Float16 Ss[M_BLK * SS_STR];     // 33,280 B (unnormalized exp)
  __shared__ float    Red[4 * 16 * 16];       //  4,096 B (split-K reduction)
  __shared__ float    MaxRed[8 * 16];         //    512 B (per-wave row maxima)
  __shared__ float    Linv[16];               //     64 B (1/rowsum)

  const int tid  = threadIdx.x;
  const int lane = tid & 31;
  const int wid  = tid >> 5;
  const int q0   = blockIdx.x * M_BLK;
  const int h    = blockIdx.y;
  const int b    = blockIdx.z;
  const size_t bh16 = (size_t)(b * H + h) * (L * E);   // halves, qh/kh/vt alike
  const int coll  = lane & 15;
  const int rbase = (lane >> 4) * 8;          // C-frag: lanes>=16 hold M=8..15

  // ---- Q A-fragments straight from global, kept in registers --------------
  v16h aq[2];
#pragma unroll
  for (int kc = 0; kc < 2; ++kc)
    aq[kc] = load_a_frag(qh + bh16 + (size_t)q0 * E, 0, E, kc * 32, lane);

  // ---- pass 1: scores = (Qs K^T)*w -> Ss (f16), running row max -----------
  float mrun[8];
#pragma unroll
  for (int r = 0; r < 8; ++r) mrun[r] = -3.0e38f;

  for (int s0 = 0; s0 < S; s0 += S_BLK) {
    v8f acc = {0.f, 0.f, 0.f, 0.f, 0.f, 0.f, 0.f, 0.f};
#pragma unroll
    for (int kc = 0; kc < 2; ++kc) {    // B-frag direct from global Kh
      v16h bf = load_b_frag(kh + bh16 + (size_t)s0 * E, wid * 16, E, kc * 32, lane);
      acc = __builtin_amdgcn_wmma_f32_16x16x32_f16(false, aq[kc], false, bf,
                                                   (short)0, acc, false, false);
    }
    const int colg = s0 + wid * 16 + coll;
    // one b128: wT[colg][q0+rbase .. q0+rbase+7] — this lane's 8 row weights
    v8h wv = *(const v8h*)(wt + (size_t)colg * L + q0 + rbase);
#pragma unroll
    for (int r = 0; r < 8; ++r) {
      const int row_l = rbase + r;
      const float sv  = acc[r] * (float)wv[r];
      mrun[r] = fmaxf(mrun[r], sv);
      Ss[row_l * SS_STR + colg] = (_Float16)sv;
    }
  }

  // cross-lane (16 columns within half-wave) max, publish per wave
#pragma unroll
  for (int r = 0; r < 8; ++r) {
    float m = mrun[r];
    m = swz_max<1>(m); m = swz_max<2>(m); m = swz_max<4>(m); m = swz_max<8>(m);
    if (coll == 0) MaxRed[wid * 16 + rbase + r] = m;
  }

  // ---- pass 2: per-row exp(x - max), keep unnormalized, record 1/sum ------
  __syncthreads();
#pragma unroll
  for (int rr = 0; rr < 2; ++rr) {
    const int row = wid * 2 + rr;
    float m = -3.0e38f;
#pragma unroll
    for (int j = 0; j < 8; ++j) m = fmaxf(m, MaxRed[j * 16 + row]);

    _Float16* sr = Ss + row * SS_STR;
    float sum = 0.f;
#pragma unroll
    for (int c = 0; c < 4; ++c) {       // 32 lanes x 8 halves x 4 = 1024
      _Float16* p = sr + c * 256 + lane * 8;
      v8h hx = *(const v8h*)p;
      v8h he;
#pragma unroll
      for (int u = 0; u < 8; ++u) {
        const float e = __expf((float)hx[u] - m);
        sum += e;
        he[u] = (_Float16)e;
      }
      *(v8h*)p = he;
    }
    sum = swz_add<1>(sum); sum = swz_add<2>(sum); sum = swz_add<4>(sum);
    sum = swz_add<8>(sum); sum = swz_add<16>(sum);
    if (lane == 0) Linv[row] = 1.0f / sum;
  }
  __syncthreads();

  // per-lane row normalizer for pass-3 A-fragments (A-frag lane row = lane&15)
  const _Float16 hinv = (_Float16)Linv[coll];

  // ---- pass 3: O = A V (split-K: wave = (dcol 0..3, shalf 0..1)) ----------
  const int dcol  = wid & 3;
  const int shalf = wid >> 2;
  v8f oacc = {0.f, 0.f, 0.f, 0.f, 0.f, 0.f, 0.f, 0.f};
  for (int s0 = 0; s0 < S; s0 += S_BLK) {
#pragma unroll
    for (int kc = 0; kc < 2; ++kc) {
      const int kb = s0 + shalf * 64 + kc * 32;
      v16h a  = load_a_frag(Ss, 0, SS_STR, kb, lane) * hinv;     // fold 1/sum
      v16h bf = load_b_frag(vt + bh16, dcol * 16, S, kb, lane);  // direct global
      oacc = __builtin_amdgcn_wmma_f32_16x16x32_f16(false, a, false, bf,
                                                    (short)0, oacc, false, false);
    }
  }

  // ---- split-K reduction + store ------------------------------------------
  if (shalf == 1) {
#pragma unroll
    for (int r = 0; r < 8; ++r) Red[(dcol * 16 + rbase + r) * 16 + coll] = oacc[r];
  }
  __syncthreads();
  if (shalf == 0) {
#pragma unroll
    for (int r = 0; r < 8; ++r) {
      const int row_l = rbase + r;
      const float o = oacc[r] + Red[(dcol * 16 + row_l) * 16 + coll];
      out[((size_t)b * L + (q0 + row_l)) * (H * D) + (size_t)h * D + dcol * 16 + coll] = o;
    }
  }
}

// ---------------------------------------------------------------------------
extern "C" void kernel_launch(void* const* d_in, const int* in_sizes, int n_in,
                              void* d_out, int out_size, void* d_ws, size_t ws_size,
                              hipStream_t stream) {
  using namespace cfg;
  const float* q   = (const float*)d_in[0];  // [B,L,H,E]
  const float* k   = (const float*)d_in[1];  // [B,S,H,E]
  const float* v   = (const float*)d_in[2];  // [B,S,H,D]
  const float* Wm  = (const float*)d_in[3];  // [1,1,L,S]
  const float* tau = (const float*)d_in[4];  // [1,1,L,1]
  // d_in[5] = attn_mask (unused, mask_flag=False)

  _Float16* wt  = (_Float16*)d_ws;                               //  2 MB [S,L]
  _Float16* qh  = (_Float16*)((char*)d_ws + (4u << 20));         // 16 MB
  _Float16* kh  = qh + PLANE;                                    // 16 MB
  _Float16* vt  = kh + PLANE;                                    // 16 MB
  float*    out = (float*)d_out;                                 // [B,L,H,D] f32

  wprep_kernel<<<L, 256, 0, stream>>>(Wm, tau, wt);
  dim3 cgrid((unsigned)(PLANE / 4 / 256), 3);
  cvt_kernel<<<cgrid, 256, 0, stream>>>(q, k, v, qh, kh, vt);
  dim3 grid(L / M_BLK, H, B);
  attn_kernel<<<grid, 256, 0, stream>>>(qh, kh, vt, wt, out);
}
